// ANR_ARL_23407571763990
// MI455X (gfx1250) — matrix operand order repaired and checked by reading, not compiled, not run on hardware
//
#include <hip/hip_runtime.h>

typedef __attribute__((ext_vector_type(16))) _Float16 v16h;
typedef __attribute__((ext_vector_type(8)))  float    v8f;

#define Bb 256
#define Ll 500
#define Ee 300
#define Hh 10
#define Kk 5
#define NCOL 64          // padded K*H (50 -> 64, 4 N-tiles of 16)
#define EPAD 320         // padded E (300 -> 320, 10 K-chunks of 32)
#define PROJ_ROWS 502    // l = -1 .. 500 (zero-padded window rows)

// LDS carve (bytes)
#define OFF_BST   0
#define SZ_BST    (NCOL * EPAD * 2)            // 40960  : aspProj^T as f16 [NCOL][EPAD]
#define OFF_PROJ  (OFF_BST + SZ_BST)
#define SZ_PROJ   (PROJ_ROWS * NCOL * 4)       // 128512 : proj f32 [502][64]
#define OFF_SC    (OFF_PROJ + SZ_PROJ)
#define SZ_SC     (Kk * 512 * 4)               // 10240  : scores/attn [5][512]
#define OFF_WE    (OFF_SC + SZ_SC)
#define SZ_WE     (160 * 4)                    // aspEmbed staging
#define OFF_PART  (OFF_WE + SZ_WE)
#define SZ_PART   (8 * 64 * 4)                 // rep partials
#define SMEM_BYTES (OFF_PART + SZ_PART)        // ~182 KB < 320 KB WGP LDS

struct Raw4 { float4 f0, f1, f2, f3; };

__device__ __forceinline__ Raw4 loadChunk(const float* __restrict__ rp, const int e0) {
  Raw4 r;
  r.f0 = *reinterpret_cast<const float4*>(rp + e0);
  r.f1 = *reinterpret_cast<const float4*>(rp + e0 + 4);
  r.f2 = *reinterpret_cast<const float4*>(rp + e0 + 8);
  r.f3 = *reinterpret_cast<const float4*>(rp + e0 + 12);
  return r;
}

__device__ __forceinline__ v16h cvtRaw(const Raw4& r) {
  v16h a;
  a[0]  = (_Float16)r.f0.x; a[1]  = (_Float16)r.f0.y;
  a[2]  = (_Float16)r.f0.z; a[3]  = (_Float16)r.f0.w;
  a[4]  = (_Float16)r.f1.x; a[5]  = (_Float16)r.f1.y;
  a[6]  = (_Float16)r.f1.z; a[7]  = (_Float16)r.f1.w;
  a[8]  = (_Float16)r.f2.x; a[9]  = (_Float16)r.f2.y;
  a[10] = (_Float16)r.f2.z; a[11] = (_Float16)r.f2.w;
  a[12] = (_Float16)r.f3.x; a[13] = (_Float16)r.f3.y;
  a[14] = (_Float16)r.f3.z; a[15] = (_Float16)r.f3.w;
  return a;
}

__global__ __launch_bounds__(512)
void anr_arl_fused(const float* __restrict__ review,   // (B,L,E)
                   const float* __restrict__ aspProj,  // (K,E,H)
                   const float* __restrict__ aspEmbed, // (K,3H)
                   float* __restrict__ out) {          // attn (B,K,L) ++ rep (B,K,H)
  extern __shared__ char smem[];
  _Float16* BsT   = (_Float16*)(smem + OFF_BST);
  float*    projL = (float*)(smem + OFF_PROJ);
  float*    scL   = (float*)(smem + OFF_SC);
  float*    wEl   = (float*)(smem + OFF_WE);
  float*    part  = (float*)(smem + OFF_PART);

  const int t    = threadIdx.x;
  const int b    = blockIdx.x;
  const int wave = t >> 5;
  const int lane = t & 31;
  const int hi   = lane >> 4;   // half-wave select
  const int ln   = lane & 15;

  // ---------------- prologue: stage B^T (f16), aspEmbed, zero pad rows -------
  for (int idx = t; idx < NCOL * EPAD; idx += 512) {
    const int c = idx / EPAD, e = idx % EPAD;
    float v = 0.0f;
    if (c < Kk * Hh && e < Ee) {
      const int k = c / Hh, h = c % Hh;
      v = aspProj[(k * Ee + e) * Hh + h];
    }
    BsT[c * EPAD + e] = (_Float16)v;
  }
  if (t < Kk * 3 * Hh) wEl[t] = aspEmbed[t];
  if (t < NCOL) {
    projL[t] = 0.0f;                              // l = -1 pad row
    projL[(PROJ_ROWS - 1) * NCOL + t] = 0.0f;     // l = 500 pad row
  }
  __syncthreads();

  // ---------------- projection GEMM: proj = review[b] x B  (WMMA f16) -------
  // M = 512 (32 tiles), N = 64 (4 tiles), Kdim = 320 (10 chunks of 32)
  const size_t rowBase = (size_t)b * Ll;
  for (int mt = wave; mt < 32; mt += 16) {
    v8f acc0 = {}, acc1 = {}, acc2 = {}, acc3 = {};
    // A layout: M = lane&15.  Clamp OOB rows to 499: their accumulator rows
    // are discarded by the l<500 store guard, so no per-lane predication.
    const int row   = mt * 16 + ln;
    const int rowC  = row < Ll ? row : (Ll - 1);
    const float* rp = review + (rowBase + rowC) * Ee;

    // prefetch the next M-tile's rows while this one computes
    if (mt + 16 < 32) {
      const int row2  = (mt + 16) * 16 + ln;
      const int rowC2 = row2 < Ll ? row2 : (Ll - 1);
      __builtin_prefetch(review + (rowBase + rowC2) * Ee, 0, 3);
    }

    // --- software-pipelined chunks 0..8: next chunk's loads in flight while
    // --- current chunk converts + WMMAs (keeps 4 b128 loads outstanding)
    Raw4 cur = loadChunk(rp, hi * 16);
#pragma unroll
    for (int ec = 0; ec < 9; ++ec) {
      Raw4 nxt;
      if (ec < 8) nxt = loadChunk(rp, (ec + 1) * 32 + hi * 16);

      const v16h a = cvtRaw(cur);
      const int koffB = ec * 32 + hi * 16;        // B: 16 contiguous K per lane
      const v16h b0 = *(const v16h*)&BsT[(0 * 16 + ln) * EPAD + koffB];
      const v16h b1 = *(const v16h*)&BsT[(1 * 16 + ln) * EPAD + koffB];
      const v16h b2 = *(const v16h*)&BsT[(2 * 16 + ln) * EPAD + koffB];
      const v16h b3 = *(const v16h*)&BsT[(3 * 16 + ln) * EPAD + koffB];
      acc0 = __builtin_amdgcn_wmma_f32_16x16x32_f16(false, a, false, b0, (short)0, acc0, false, false);
      acc1 = __builtin_amdgcn_wmma_f32_16x16x32_f16(false, a, false, b1, (short)0, acc1, false, false);
      acc2 = __builtin_amdgcn_wmma_f32_16x16x32_f16(false, a, false, b2, (short)0, acc2, false, false);
      acc3 = __builtin_amdgcn_wmma_f32_16x16x32_f16(false, a, false, b3, (short)0, acc3, false, false);

      if (ec < 8) cur = nxt;
    }

    { // tail chunk ec=9: e = 288..319; K=300..319 always OOB -> zero;
      // hi=0 half reads e=288..303 (first 12 valid), hi=1 half e=304..319
      // (all OOB, zero).  Matching B columns are zero-padded too.
      v16h a = {};
      if (hi == 0) {
        const float4 f0 = *reinterpret_cast<const float4*>(rp + 288);
        const float4 f1 = *reinterpret_cast<const float4*>(rp + 292);
        const float4 f2 = *reinterpret_cast<const float4*>(rp + 296);
        a[0] = (_Float16)f0.x; a[1] = (_Float16)f0.y;
        a[2] = (_Float16)f0.z; a[3] = (_Float16)f0.w;
        a[4] = (_Float16)f1.x; a[5] = (_Float16)f1.y;
        a[6] = (_Float16)f1.z; a[7] = (_Float16)f1.w;
        a[8] = (_Float16)f2.x; a[9] = (_Float16)f2.y;
        a[10] = (_Float16)f2.z; a[11] = (_Float16)f2.w;
      }
      const int koffB = 288 + hi * 16;
      const v16h b0 = *(const v16h*)&BsT[(0 * 16 + ln) * EPAD + koffB];
      const v16h b1 = *(const v16h*)&BsT[(1 * 16 + ln) * EPAD + koffB];
      const v16h b2 = *(const v16h*)&BsT[(2 * 16 + ln) * EPAD + koffB];
      const v16h b3 = *(const v16h*)&BsT[(3 * 16 + ln) * EPAD + koffB];
      acc0 = __builtin_amdgcn_wmma_f32_16x16x32_f16(false, a, false, b0, (short)0, acc0, false, false);
      acc1 = __builtin_amdgcn_wmma_f32_16x16x32_f16(false, a, false, b1, (short)0, acc1, false, false);
      acc2 = __builtin_amdgcn_wmma_f32_16x16x32_f16(false, a, false, b2, (short)0, acc2, false, false);
      acc3 = __builtin_amdgcn_wmma_f32_16x16x32_f16(false, a, false, b3, (short)0, acc3, false, false);
    }

    // D layout: c[v] holds M = mt*16 + v + hi*8, N = nt*16 + ln
#pragma unroll
    for (int v = 0; v < 8; ++v) {
      const int l = mt * 16 + v + hi * 8;
      if (l < Ll) {
        float* prow = &projL[(l + 1) * NCOL];
        prow[0 * 16 + ln] = acc0[v];
        prow[1 * 16 + ln] = acc1[v];
        prow[2 * 16 + ln] = acc2[v];
        prow[3 * 16 + ln] = acc3[v];
      }
    }
  }
  __syncthreads();

  // ---------------- window-3 attention scores (f = h*3 + offset) ------------
  for (int l = t; l < Ll; l += 512) {
#pragma unroll
    for (int k = 0; k < Kk; ++k) {
      float s = 0.0f;
#pragma unroll
      for (int h = 0; h < Hh; ++h) {
        const int c = k * Hh + h;
        s += projL[(l + 0) * NCOL + c] * wEl[k * 30 + h * 3 + 0];  // l-1
        s += projL[(l + 1) * NCOL + c] * wEl[k * 30 + h * 3 + 1];  // l
        s += projL[(l + 2) * NCOL + c] * wEl[k * 30 + h * 3 + 2];  // l+1
      }
      scL[k * 512 + l] = s;
    }
  }
  __syncthreads();

  // ---------------- softmax over L, one wave per aspect k -------------------
  if (wave < Kk) {
    const int k = wave;
    float m = -3.402823466e38f;
    for (int l = lane; l < Ll; l += 32) m = fmaxf(m, scL[k * 512 + l]);
#pragma unroll
    for (int off = 16; off > 0; off >>= 1) m = fmaxf(m, __shfl_xor(m, off, 32));
    float s = 0.0f;
    for (int l = lane; l < Ll; l += 32) {
      const float ex = __expf(scL[k * 512 + l] - m);
      scL[k * 512 + l] = ex;
      s += ex;
    }
#pragma unroll
    for (int off = 16; off > 0; off >>= 1) s += __shfl_xor(s, off, 32);
    const float inv = 1.0f / s;
    for (int l = lane; l < Ll; l += 32) scL[k * 512 + l] *= inv;
  }
  __syncthreads();

  // ---------------- write attn (B,K,L) --------------------------------------
  float* attnOut = out;
  float* repOut  = out + (size_t)Bb * Kk * Ll;
  for (int idx = t; idx < Kk * Ll; idx += 512) {
    const int k = idx / Ll, l = idx % Ll;
    attnOut[((size_t)b * Kk + k) * Ll + l] = scL[k * 512 + l];
  }

  // ---------------- rep[k,h] = sum_l proj * attn -----------------------------
  {
    const int c = t & 63;     // column = k*10+h
    const int g = t >> 6;     // 8 l-strided groups
    float acc = 0.0f;
    if (c < Kk * Hh) {
      const int k = c / Hh;
      for (int l = g; l < Ll; l += 8)
        acc += projL[(l + 1) * NCOL + c] * scL[k * 512 + l];
    }
    part[g * 64 + c] = acc;
  }
  __syncthreads();
  if (t < Kk * Hh) {
    float r = 0.0f;
#pragma unroll
    for (int g = 0; g < 8; ++g) r += part[g * 64 + t];
    repOut[(size_t)b * Kk * Hh + t] = r;
  }
}

extern "C" void kernel_launch(void* const* d_in, const int* in_sizes, int n_in,
                              void* d_out, int out_size, void* d_ws, size_t ws_size,
                              hipStream_t stream) {
  const float* review  = (const float*)d_in[0];
  const float* aspProj = (const float*)d_in[1];
  const float* aspEmb  = (const float*)d_in[2];
  float* out = (float*)d_out;
  (void)in_sizes; (void)n_in; (void)out_size; (void)d_ws; (void)ws_size;
  hipLaunchKernelGGL(anr_arl_fused, dim3(Bb), dim3(512), SMEM_BYTES, stream,
                     review, aspProj, aspEmb, out);
}
